// CrystalGraphConv_14757507629563
// MI455X (gfx1250) — matrix operand order repaired and checked by reading, not compiled, not run on hardware
//
#include <hip/hip_runtime.h>
#include <hip/hip_bf16.h>

#define NN    50000
#define EE    1600000
#define GG    256
#define ATOMF 92
#define EDGEF 35
#define DD    64
#define HH    128
#define LL    3
#define NTILES (EE / 16)            // 100000 (EE divisible by 16)
#define NSUPER (NTILES / 2)         // 50000 supertiles of 32 edges
#define BFRAG (2 * 4 * 4 * 32 * 16) // packed B halves per layer = 16384
#define EDGE_BLOCKS 1536

typedef __attribute__((ext_vector_type(16))) _Float16 v16h;
typedef __attribute__((ext_vector_type(8)))  _Float16 v8h;
typedef __attribute__((ext_vector_type(8)))  float    v8f;

// fast device activations: v_exp_f32 / v_log_f32 / v_rcp_f32 (TRANS ops)
__device__ __forceinline__ float softplus_f(float x) {
  return fmaxf(x, 0.f) + __logf(1.f + __expf(-fabsf(x)));
}
__device__ __forceinline__ float sigmoid_f(float x) {
  return __builtin_amdgcn_rcpf(1.f + __expf(-x));
}

// ---------------------------------------------------------------------------
// Pack W_f / W_s [L,99,64] f32 -> f16 B fragments in WMMA layout:
//   Bpack[l][mat][kt][nt][lane][j]  where  k = kt*32 + (lane>>4)*8 + (j&7) + (j>>3)*16,
//                                         n = nt*16 + (lane&15),  k>=99 -> 0
// ---------------------------------------------------------------------------
__global__ void pack_weights_kernel(const float* __restrict__ W_f,
                                    const float* __restrict__ W_s,
                                    _Float16* __restrict__ Bpack) {
  int idx = blockIdx.x * blockDim.x + threadIdx.x;
  const int total = LL * BFRAG;
  if (idx >= total) return;
  int j    = idx & 15;
  int lane = (idx >> 4) & 31;
  int nt   = (idx >> 9) & 3;
  int kt   = (idx >> 11) & 3;
  int mat  = (idx >> 13) & 1;
  int l    = idx >> 14;
  int k = kt * 32 + (lane >> 4) * 8 + (j & 7) + (j >> 3) * 16;
  int n = nt * 16 + (lane & 15);
  const float* W = mat ? W_s : W_f;
  float v = (k < DD + EDGEF) ? W[((size_t)l * (DD + EDGEF) + k) * DD + n] : 0.f;
  Bpack[idx] = (_Float16)v;
}

// h = x @ W_emb + b_emb
__global__ void embed_kernel(const float* __restrict__ x, const float* __restrict__ W,
                             const float* __restrict__ b, float* __restrict__ h) {
  int idx = blockIdx.x * blockDim.x + threadIdx.x;
  if (idx >= NN * DD) return;
  int n = idx >> 6, d = idx & 63;
  float acc = b[d];
  const float* xr = x + (size_t)n * ATOMF;
#pragma unroll 4
  for (int k = 0; k < ATOMF; ++k) acc += xr[k] * W[k * DD + d];
  h[idx] = acc;
}

__global__ void zero_kernel(float* __restrict__ p, int n) {
  int i = blockIdx.x * blockDim.x + threadIdx.x;
  if (i < n) p[i] = 0.f;
}

// Build one 8-half run of z[m][kstart..kstart+7] in the A-fragment layout.
__device__ __forceinline__ v8h load_z8(const float* __restrict__ h,
                                       const float* __restrict__ ea,
                                       int dstm, int srcm, int e, int kstart) {
  v8h r;
  if (kstart < DD) {
    const float4* pd = (const float4*)(h + (size_t)dstm * DD + kstart);
    const float4* ps = (const float4*)(h + (size_t)srcm * DD + kstart);
    float4 d0 = pd[0], d1 = pd[1];
    float4 s0 = ps[0], s1 = ps[1];
    r[0] = (_Float16)(0.5f * (d0.x + s0.x));
    r[1] = (_Float16)(0.5f * (d0.y + s0.y));
    r[2] = (_Float16)(0.5f * (d0.z + s0.z));
    r[3] = (_Float16)(0.5f * (d0.w + s0.w));
    r[4] = (_Float16)(0.5f * (d1.x + s1.x));
    r[5] = (_Float16)(0.5f * (d1.y + s1.y));
    r[6] = (_Float16)(0.5f * (d1.z + s1.z));
    r[7] = (_Float16)(0.5f * (d1.w + s1.w));
  } else {
#pragma unroll
    for (int i = 0; i < 8; ++i) {
      int k = kstart - DD + i;
      // edge_attr is streamed once per layer: keep it out of L2 (non-temporal)
      r[i] = (k < EDGEF) ? (_Float16)__builtin_nontemporal_load(&ea[(size_t)e * EDGEF + k])
                         : (_Float16)0.f;
    }
  }
  return r;
}

__device__ __forceinline__ v16h cat16(v8h lo, v8h hi) {
  return __builtin_shufflevector(lo, hi, 0, 1, 2, 3, 4, 5, 6, 7,
                                 8, 9, 10, 11, 12, 13, 14, 15);
}

// ---------------------------------------------------------------------------
// Fused per-edge message + gated activation + scatter-add.
// Persistent waves; each wave iteration processes a 32-edge supertile
// (two 16x16x128 A tiles sharing every B fragment) ->
// 64 x v_wmma_f32_16x16x32_f16 per supertile, half the LDS reads per MAC.
// __launch_bounds__(.,1): allow full VGPR allocation, no hot-loop spills.
// ---------------------------------------------------------------------------
__global__ void __launch_bounds__(256, 1)
edge_msg_kernel(const float* __restrict__ h,
                const int* __restrict__ edge_index,
                const float* __restrict__ edge_attr,
                const _Float16* __restrict__ Bpack,  // this layer
                const float* __restrict__ b_f,       // [64] slice
                const float* __restrict__ b_s,       // [64] slice
                float* __restrict__ agg) {
  __shared__ __align__(32) _Float16 Bsh[BFRAG];  // 32 KB, staged once per block
  const int tid = threadIdx.x;
  for (int i = tid; i < BFRAG / 8; i += blockDim.x)
    ((float4*)Bsh)[i] = ((const float4*)Bpack)[i];
  __syncthreads();

  const int lane = tid & 31;
  const int half = lane >> 4;  // 0: K%32 in {0..7,16..23}; 1: {8..15,24..31}
  const int lm   = lane & 15;
  const int waveId     = blockIdx.x * (blockDim.x >> 5) + (tid >> 5);
  const int waveStride = gridDim.x * (blockDim.x >> 5);

  for (int s = waveId; s < NSUPER; s += waveStride) {
    const int e0 = s * 32;

    // speculative prefetch of next supertile's edge_attr rows
    const int sn = s + waveStride;
    if (sn < NSUPER)
      __builtin_prefetch(edge_attr + (size_t)(sn * 32 + lane) * EDGEF, 0, 0);

    // ---- A fragments: rows m = lm of the two z tiles (ISA 16-bit A layout)
    const int eA0 = e0 + lm;
    const int eA1 = e0 + 16 + lm;
    const int sr0 = edge_index[eA0], dr0 = edge_index[EE + eA0];
    const int sr1 = edge_index[eA1], dr1 = edge_index[EE + eA1];
    v16h A0[4], A1[4];
#pragma unroll
    for (int kt = 0; kt < 4; ++kt) {
      const int k0 = kt * 32 + half * 8;
      A0[kt] = cat16(load_z8(h, edge_attr, dr0, sr0, eA0, k0),
                     load_z8(h, edge_attr, dr0, sr0, eA0, k0 + 16));
      A1[kt] = cat16(load_z8(h, edge_attr, dr1, sr1, eA1, k0),
                     load_z8(h, edge_attr, dr1, sr1, eA1, k0 + 16));
    }

    // dst rows for the 8 accumulator rows this lane owns (M = half*8 + r)
    int drow0[8], drow1[8];
#pragma unroll
    for (int r = 0; r < 8; ++r) {
      drow0[r] = edge_index[EE + e0 + half * 8 + r];
      drow1[r] = edge_index[EE + e0 + 16 + half * 8 + r];
    }

#pragma unroll
    for (int nt = 0; nt < 4; ++nt) {
      v8f cf0 = {}, cs0 = {}, cf1 = {}, cs1 = {};
#pragma unroll
      for (int kt = 0; kt < 4; ++kt) {
        const v8h* bf = (const v8h*)&Bsh[(((0 * 4 + kt) * 4 + nt) * 32 + lane) * 16];
        v16h Bf = cat16(bf[0], bf[1]);
        cf0 = __builtin_amdgcn_wmma_f32_16x16x32_f16(false, A0[kt], false, Bf,
                                                     (short)0, cf0, false, false);
        cf1 = __builtin_amdgcn_wmma_f32_16x16x32_f16(false, A1[kt], false, Bf,
                                                     (short)0, cf1, false, false);
        const v8h* bs = (const v8h*)&Bsh[(((1 * 4 + kt) * 4 + nt) * 32 + lane) * 16];
        v16h Bs = cat16(bs[0], bs[1]);
        cs0 = __builtin_amdgcn_wmma_f32_16x16x32_f16(false, A0[kt], false, Bs,
                                                     (short)0, cs0, false, false);
        cs1 = __builtin_amdgcn_wmma_f32_16x16x32_f16(false, A1[kt], false, Bs,
                                                     (short)0, cs1, false, false);
      }
      const int n = nt * 16 + lm;
      const float bfv = b_f[n];
      const float bsv = b_s[n];
#pragma unroll
      for (int r = 0; r < 8; ++r) {
        float m0 = sigmoid_f(cf0[r] + bfv) * softplus_f(cs0[r] + bsv);
        atomicAdd(&agg[(size_t)drow0[r] * DD + n], m0);
      }
#pragma unroll
      for (int r = 0; r < 8; ++r) {
        float m1 = sigmoid_f(cf1[r] + bfv) * softplus_f(cs1[r] + bsv);
        atomicAdd(&agg[(size_t)drow1[r] * DD + n], m1);
      }
    }
  }
}

// per-channel sum & sumsq of agg -> stats[0..63], stats[64..127]
__global__ void bn_stats_kernel(const float* __restrict__ agg, float* __restrict__ stats) {
  __shared__ float ssum[256];
  __shared__ float ssq[256];
  const int d = threadIdx.x & 63;
  const int g = threadIdx.x >> 6;  // 0..3
  float s = 0.f, q = 0.f;
  for (int n = blockIdx.x * 4 + g; n < NN; n += gridDim.x * 4) {
    float v = agg[(size_t)n * DD + d];
    s += v;
    q += v * v;
  }
  ssum[threadIdx.x] = s;
  ssq[threadIdx.x] = q;
  __syncthreads();
  if (g == 0) {
    s = ssum[d] + ssum[64 + d] + ssum[128 + d] + ssum[192 + d];
    q = ssq[d] + ssq[64 + d] + ssq[128 + d] + ssq[192 + d];
    atomicAdd(&stats[d], s);
    atomicAdd(&stats[64 + d], q);
  }
}

// h = gamma*(agg-mu)*rsqrt(var+eps) + beta + h   (biased variance)
__global__ void bn_apply_kernel(const float* __restrict__ agg, const float* __restrict__ stats,
                                const float* __restrict__ gamma, const float* __restrict__ beta,
                                float* __restrict__ h) {
  int idx = blockIdx.x * blockDim.x + threadIdx.x;
  if (idx >= NN * DD) return;
  int d = idx & 63;
  float mu  = stats[d] * (1.0f / NN);
  float var = stats[64 + d] * (1.0f / NN) - mu * mu;
  float inv = rsqrtf(var + 1e-5f);
  h[idx] = gamma[d] * (agg[idx] - mu) * inv + beta[d] + h[idx];
}

// out = h @ W_l1 + b_l1
__global__ void lin1_kernel(const float* __restrict__ h, const float* __restrict__ W,
                            const float* __restrict__ b, float* __restrict__ out) {
  int idx = blockIdx.x * blockDim.x + threadIdx.x;
  if (idx >= NN * DD) return;
  int n = idx >> 6, d = idx & 63;
  float acc = b[d];
  const float* hr = h + (size_t)n * DD;
#pragma unroll 8
  for (int k = 0; k < DD; ++k) acc += hr[k] * W[k * DD + d];
  out[idx] = acc;
}

__global__ void pool_kernel(const float* __restrict__ h2, const int* __restrict__ batch,
                            float* __restrict__ pool, float* __restrict__ cnt) {
  int idx = blockIdx.x * blockDim.x + threadIdx.x;
  if (idx >= NN * DD) return;
  int n = idx >> 6, d = idx & 63;
  int g = batch[n];
  atomicAdd(&pool[g * DD + d], h2[idx]);
  if (d == 0) atomicAdd(&cnt[g], 1.0f);
}

// g = softplus(pooled/cnt); g = softplus(g@W_fc+b_fc); out = g@W_out + b_out
__global__ void head_kernel(const float* __restrict__ pool, const float* __restrict__ cnt,
                            const float* __restrict__ W_fc, const float* __restrict__ b_fc,
                            const float* __restrict__ W_out, const float* __restrict__ b_out,
                            float* __restrict__ out) {
  int g = blockIdx.x * blockDim.x + threadIdx.x;
  if (g >= GG) return;
  float v[DD];
  float c = fmaxf(cnt[g], 1.0f);
#pragma unroll
  for (int d = 0; d < DD; ++d) v[d] = softplus_f(pool[g * DD + d] / c);
  float acc = 0.f;
  for (int j = 0; j < HH; ++j) {
    float a = b_fc[j];
#pragma unroll
    for (int d = 0; d < DD; ++d) a += v[d] * W_fc[d * HH + j];
    acc += softplus_f(a) * W_out[j];
  }
  out[g] = acc + b_out[0];
}

extern "C" void kernel_launch(void* const* d_in, const int* in_sizes, int n_in,
                              void* d_out, int out_size, void* d_ws, size_t ws_size,
                              hipStream_t stream) {
  const float* x          = (const float*)d_in[0];
  const int*   edge_index = (const int*)d_in[1];
  const float* edge_attr  = (const float*)d_in[2];
  const int*   batch      = (const int*)d_in[3];
  const float* W_emb      = (const float*)d_in[4];
  const float* b_emb      = (const float*)d_in[5];
  const float* W_f        = (const float*)d_in[6];
  const float* b_f        = (const float*)d_in[7];
  const float* W_s        = (const float*)d_in[8];
  const float* b_s        = (const float*)d_in[9];
  const float* bn_gamma   = (const float*)d_in[10];
  const float* bn_beta    = (const float*)d_in[11];
  const float* W_l1       = (const float*)d_in[12];
  const float* b_l1       = (const float*)d_in[13];
  const float* W_fc       = (const float*)d_in[14];
  const float* b_fc       = (const float*)d_in[15];
  const float* W_out      = (const float*)d_in[16];
  const float* b_out      = (const float*)d_in[17];
  float* out = (float*)d_out;

  // workspace layout (floats); Bpack offset is 16B-aligned by construction
  float* h     = (float*)d_ws;
  float* agg   = h + (size_t)NN * DD;
  float* stats = agg + (size_t)NN * DD;
  float* pool  = stats + 2 * DD;
  float* cnt   = pool + (size_t)GG * DD;
  _Float16* Bpack = (_Float16*)(cnt + GG);

  pack_weights_kernel<<<(LL * BFRAG + 255) / 256, 256, 0, stream>>>(W_f, W_s, Bpack);
  embed_kernel<<<(NN * DD + 255) / 256, 256, 0, stream>>>(x, W_emb, b_emb, h);

  for (int l = 0; l < LL; ++l) {
    zero_kernel<<<(NN * DD + 255) / 256, 256, 0, stream>>>(agg, NN * DD);
    zero_kernel<<<1, 256, 0, stream>>>(stats, 2 * DD);
    edge_msg_kernel<<<EDGE_BLOCKS, 256, 0, stream>>>(
        h, edge_index, edge_attr, Bpack + (size_t)l * BFRAG,
        b_f + l * DD, b_s + l * DD, agg);
    bn_stats_kernel<<<512, 256, 0, stream>>>(agg, stats);
    bn_apply_kernel<<<(NN * DD + 255) / 256, 256, 0, stream>>>(
        agg, stats, bn_gamma + l * DD, bn_beta + l * DD, h);
  }

  lin1_kernel<<<(NN * DD + 255) / 256, 256, 0, stream>>>(h, W_l1, b_l1, agg);
  zero_kernel<<<(GG * DD + GG + 255) / 256, 256, 0, stream>>>(pool, GG * DD + GG);
  pool_kernel<<<(NN * DD + 255) / 256, 256, 0, stream>>>(agg, batch, pool, cnt);
  head_kernel<<<(GG + 127) / 128, 128, 0, stream>>>(pool, cnt, W_fc, b_fc, W_out, b_out, out);
}
